// TPUBalancedMoE_19756849562328
// MI455X (gfx1250) — matrix operand order repaired and checked by reading, not compiled, not run on hardware
//
#include <hip/hip_runtime.h>
#include <hip/hip_bf16.h>

// Problem constants (fixed by the reference)
#define T_TOK 8192      // B*S tokens
#define HDIM  1024
#define DDIM  2048
#define NEXP  8
#define PADR  32                          // expert region padding = M block tile
#define CAPR  (2 * T_TOK + NEXP * PADR)   // 16640 flat expert-row slots
#define CAPT  (CAPR / PADR)               // 520 row tiles (32 rows each)
#define ZROWN 2048                        // zero dummy row length (>= max(H, D))

typedef __attribute__((ext_vector_type(16))) __bf16 v16bf;
typedef __attribute__((ext_vector_type(8)))  __bf16 v8bf;
typedef __attribute__((ext_vector_type(8)))  float  v8f;

// ---------------------------------------------------------------- init
__global__ __launch_bounds__(256) void zero_out_kernel(float* __restrict__ out, int n) {
    int i = blockIdx.x * 256 + threadIdx.x;
    if (i < n) out[i] = 0.0f;
}

__global__ __launch_bounds__(256) void init_meta_kernel(int* __restrict__ row_token,
                                                        int* __restrict__ tile_expert,
                                                        int* __restrict__ counts,
                                                        int* __restrict__ cursor,
                                                        __bf16* __restrict__ zrow) {
    int i = blockIdx.x * 256 + threadIdx.x;
    if (i < CAPR) row_token[i] = -1;
    if (i < CAPT) tile_expert[i] = -1;
    if (i < ZROWN) zrow[i] = (__bf16)0.0f;
    if (i < NEXP) { counts[i] = 0; cursor[i] = 0; }
}

// ---------------------------------------------------------------- fp32 -> bf16 conversions
__global__ __launch_bounds__(256) void convert_x_kernel(const float* __restrict__ x,
                                                        __bf16* __restrict__ xb, int n) {
    int i = blockIdx.x * 256 + threadIdx.x;
    if (i < n) xb[i] = (__bf16)x[i];
}

// W[e][k][n] (K x N row-major) -> WT[e][n][k] bf16, LDS-tiled so both sides coalesce.
__global__ __launch_bounds__(256) void transpose_bf16_kernel(const float* __restrict__ W,
                                                             __bf16* __restrict__ WT,
                                                             int K, int N) {
    __shared__ float tile[32][33];
    const int e = blockIdx.z;
    const float* Wp = W + (size_t)e * K * N;
    __bf16* WTp = WT + (size_t)e * K * N;
    const int n0 = blockIdx.x * 32, k0 = blockIdx.y * 32;
    const int tx = threadIdx.x, ty = threadIdx.y;   // (32, 8)
#pragma unroll
    for (int i = 0; i < 32; i += 8)
        tile[ty + i][tx] = Wp[(size_t)(k0 + ty + i) * N + n0 + tx];
    __syncthreads();
#pragma unroll
    for (int i = 0; i < 32; i += 8)
        WTp[(size_t)(n0 + ty + i) * K + k0 + tx] = (__bf16)tile[tx][ty + i];
}

// ---------------------------------------------------------------- routing (one wave per token)
__global__ __launch_bounds__(256) void routing_kernel(const float* __restrict__ x,
                                                      const float* __restrict__ rw,
                                                      int* __restrict__ top_e,
                                                      float* __restrict__ top_g,
                                                      int* __restrict__ counts) {
    const int wave = threadIdx.x >> 5, lane = threadIdx.x & 31;
    const int t = blockIdx.x * 8 + wave;
    if (t >= T_TOK) return;
    const float* xr = x + (size_t)t * HDIM;
    float acc[NEXP];
#pragma unroll
    for (int e = 0; e < NEXP; ++e) acc[e] = 0.0f;
    for (int k = lane; k < HDIM; k += 32) {
        float xv = xr[k];
#pragma unroll
        for (int e = 0; e < NEXP; ++e) acc[e] += xv * rw[k * NEXP + e];
    }
#pragma unroll
    for (int e = 0; e < NEXP; ++e)
#pragma unroll
        for (int off = 16; off > 0; off >>= 1) acc[e] += __shfl_xor(acc[e], off, 32);
    if (lane == 0) {
        float mx = acc[0];
#pragma unroll
        for (int e = 1; e < NEXP; ++e) mx = fmaxf(mx, acc[e]);
        float p[NEXP], s = 0.0f;
#pragma unroll
        for (int e = 0; e < NEXP; ++e) { p[e] = __expf(acc[e] - mx); s += p[e]; }
        float inv = 1.0f / s;
#pragma unroll
        for (int e = 0; e < NEXP; ++e) p[e] *= inv;
        int i0 = 0;
#pragma unroll
        for (int e = 1; e < NEXP; ++e) if (p[e] > p[i0]) i0 = e;
        int i1 = (i0 == 0) ? 1 : 0;
#pragma unroll
        for (int e = 0; e < NEXP; ++e) if (e != i0 && p[e] > p[i1]) i1 = e;
        top_e[t * 2 + 0] = i0; top_g[t * 2 + 0] = p[i0];
        top_e[t * 2 + 1] = i1; top_g[t * 2 + 1] = p[i1];
        atomicAdd(&counts[i0], 1);
        atomicAdd(&counts[i1], 1);
    }
}

// ---------------------------------------------------------------- scan: padded per-expert offsets
__global__ void scan_kernel(const int* __restrict__ counts, int* __restrict__ cursor,
                            int* __restrict__ tile_expert) {
    if (threadIdx.x == 0 && blockIdx.x == 0) {
        int off = 0;
        for (int e = 0; e < NEXP; ++e) {
            cursor[e] = off;
            int tiles = (counts[e] + PADR - 1) / PADR;   // pad each expert region to 32 rows
            int t0 = off / PADR;
            for (int tt = 0; tt < tiles; ++tt) tile_expert[t0 + tt] = e;
            off += tiles * PADR;
        }
    }
}

__global__ __launch_bounds__(256) void scatter_kernel(const int* __restrict__ top_e,
                                                      const float* __restrict__ top_g,
                                                      int* __restrict__ cursor,
                                                      int* __restrict__ row_token,
                                                      float* __restrict__ row_gate) {
    int t = blockIdx.x * 256 + threadIdx.x;
    if (t >= T_TOK) return;
#pragma unroll
    for (int k = 0; k < 2; ++k) {
        int e = top_e[t * 2 + k];
        int slot = atomicAdd(&cursor[e], 1);
        row_token[slot] = t;
        row_gate[slot]  = top_g[t * 2 + k];
    }
}

// ---------------------------------------------------------------- GEMM1: hmid = gelu(Xg @ W1[e] + b1[e])
// block = 256 thr = 8 waves; block tile 32 rows x 512 cols; wave tile 32x64 = 2 M x 4 N WMMA tiles.
// Per k-step (K=32): 4 A-chunk loads + 4 B-vector loads feed 8 independent WMMAs.
__global__ __launch_bounds__(256) void gemm1_kernel(const __bf16* __restrict__ xb,
                                                    const __bf16* __restrict__ W1T,   // [E][D][H]
                                                    const float* __restrict__ b1,     // [E][D]
                                                    __bf16* __restrict__ hmid,        // [CAPR][D]
                                                    const int* __restrict__ row_token,
                                                    const int* __restrict__ tile_expert,
                                                    const __bf16* __restrict__ zrow) {
    const int tile = blockIdx.x;
    const int e = tile_expert[tile];
    if (e < 0) return;
    const int w = threadIdx.x >> 5, lane = threadIdx.x & 31;
    const int half = lane >> 4;                 // 0: lanes 0-15, 1: lanes 16-31
    const int m    = lane & 15;
    const int r0   = tile * PADR;               // 32-row tile, single expert
    const int n0   = blockIdx.y * 512 + w * 64; // wave covers 64 N columns
    const int ka   = half * 16;                 // A per-lane K run:  ka..ka+15
    const int kb   = half * 16;                 // B per-lane K run:  kb..kb+15

    // A row pointers (pad rows -> zero dummy row; no branch in the k-loop)
    const int tok0 = row_token[r0 + m];
    const int tok1 = row_token[r0 + 16 + m];
    const __bf16* Arow0 = (tok0 >= 0) ? (xb + (size_t)tok0 * HDIM) : zrow;
    const __bf16* Arow1 = (tok1 >= 0) ? (xb + (size_t)tok1 * HDIM) : zrow;
    // One B base; the 4 N-subtiles are constant immediate offsets (j*16*H elements)
    const __bf16* Bbase = W1T + ((size_t)e * DDIM + n0 + m) * HDIM;

    v8f acc[2][4] = {};
#pragma unroll 2
    for (int k0 = 0; k0 < HDIM; k0 += 32) {
        v16bf a0, a1;
        {
            v8bf lo = *(const v8bf*)(Arow0 + k0 + ka);
            v8bf hi = *(const v8bf*)(Arow0 + k0 + ka + 8);
#pragma unroll
            for (int i = 0; i < 8; ++i) { a0[i] = lo[i]; a0[8 + i] = hi[i]; }
        }
        {
            v8bf lo = *(const v8bf*)(Arow1 + k0 + ka);
            v8bf hi = *(const v8bf*)(Arow1 + k0 + ka + 8);
#pragma unroll
            for (int i = 0; i < 8; ++i) { a1[i] = lo[i]; a1[8 + i] = hi[i]; }
        }
#pragma unroll
        for (int j = 0; j < 4; ++j) {
            v16bf b = *(const v16bf*)(Bbase + (size_t)j * 16 * HDIM + k0 + kb);
            acc[0][j] = __builtin_amdgcn_wmma_f32_16x16x32_bf16(false, a0, false, b,
                                                                (short)0, acc[0][j], false, false);
            acc[1][j] = __builtin_amdgcn_wmma_f32_16x16x32_bf16(false, a1, false, b,
                                                                (short)0, acc[1][j], false, false);
        }
    }
#pragma unroll
    for (int j = 0; j < 4; ++j) {
        const int n = n0 + j * 16 + m;
        const float bias = b1[e * DDIM + n];
#pragma unroll
        for (int i = 0; i < 2; ++i) {
#pragma unroll
            for (int r = 0; r < 8; ++r) {
                const int mr = i * 16 + r + half * 8;   // C layout: lanes>=16 hold M = r+8
                float v = acc[i][j][r] + bias;
                float g = 0.5f * v * (1.0f + tanhf(0.7978845608028654f *
                                                   (v + 0.044715f * v * v * v)));
                hmid[(size_t)(r0 + mr) * DDIM + n] = (__bf16)g;
            }
        }
    }
}

// ---------------------------------------------------------------- GEMM2: out[token] += gate*(hmid @ W2[e] + b2[e])
__global__ __launch_bounds__(256) void gemm2_kernel(const __bf16* __restrict__ hmid,   // [CAPR][D]
                                                    const __bf16* __restrict__ W2T,    // [E][H][D]
                                                    const float* __restrict__ b2,      // [E][H]
                                                    float* __restrict__ out,           // [T][H]
                                                    const int* __restrict__ row_token,
                                                    const float* __restrict__ row_gate,
                                                    const int* __restrict__ tile_expert) {
    const int tile = blockIdx.x;
    const int e = tile_expert[tile];
    if (e < 0) return;
    const int w = threadIdx.x >> 5, lane = threadIdx.x & 31;
    const int half = lane >> 4;
    const int m    = lane & 15;
    const int r0   = tile * PADR;
    const int n0   = blockIdx.y * 512 + w * 64;
    const int ka   = half * 16;
    const int kb   = half * 16;

    const __bf16* Arow0 = hmid + (size_t)(r0 + m) * DDIM;        // pad rows hold valid data
    const __bf16* Arow1 = hmid + (size_t)(r0 + 16 + m) * DDIM;
    const __bf16* Bbase = W2T + ((size_t)e * HDIM + n0 + m) * DDIM;

    v8f acc[2][4] = {};
#pragma unroll 2
    for (int k0 = 0; k0 < DDIM; k0 += 32) {
        v16bf a0, a1;
        {
            v8bf lo = *(const v8bf*)(Arow0 + k0 + ka);
            v8bf hi = *(const v8bf*)(Arow0 + k0 + ka + 8);
#pragma unroll
            for (int i = 0; i < 8; ++i) { a0[i] = lo[i]; a0[8 + i] = hi[i]; }
        }
        {
            v8bf lo = *(const v8bf*)(Arow1 + k0 + ka);
            v8bf hi = *(const v8bf*)(Arow1 + k0 + ka + 8);
#pragma unroll
            for (int i = 0; i < 8; ++i) { a1[i] = lo[i]; a1[8 + i] = hi[i]; }
        }
#pragma unroll
        for (int j = 0; j < 4; ++j) {
            v16bf b = *(const v16bf*)(Bbase + (size_t)j * 16 * DDIM + k0 + kb);
            acc[0][j] = __builtin_amdgcn_wmma_f32_16x16x32_bf16(false, a0, false, b,
                                                                (short)0, acc[0][j], false, false);
            acc[1][j] = __builtin_amdgcn_wmma_f32_16x16x32_bf16(false, a1, false, b,
                                                                (short)0, acc[1][j], false, false);
        }
    }
#pragma unroll
    for (int j = 0; j < 4; ++j) {
        const int n = n0 + j * 16 + m;
        const float bias = b2[e * HDIM + n];
#pragma unroll
        for (int i = 0; i < 2; ++i) {
#pragma unroll
            for (int r = 0; r < 8; ++r) {
                const int mr = i * 16 + r + half * 8;
                const int row = r0 + mr;
                const int tok = row_token[row];
                if (tok >= 0) {
                    float val = (acc[i][j][r] + bias) * row_gate[row];
                    atomicAdd(out + (size_t)tok * HDIM + n, val);
                }
            }
        }
    }
}

// ---------------------------------------------------------------- launch
extern "C" void kernel_launch(void* const* d_in, const int* in_sizes, int n_in,
                              void* d_out, int out_size, void* d_ws, size_t ws_size,
                              hipStream_t stream) {
    const float* x   = (const float*)d_in[0];
    // d_in[1] router_w, d_in[2] router_b: logits unused downstream -> skipped
    const float* rw  = (const float*)d_in[3];   // routing_weights [H, E]
    const float* W1  = (const float*)d_in[4];   // [E, H, D]
    const float* b1  = (const float*)d_in[5];   // [E, D]
    const float* W2  = (const float*)d_in[6];   // [E, D, H]
    const float* b2  = (const float*)d_in[7];   // [E, H]
    float* out = (float*)d_out;

    // Workspace carve-out (~153 MB total)
    char* p = (char*)d_ws;
    auto take = [&](size_t bytes) -> void* {
        void* r = (void*)p;
        p += (bytes + 255) & ~(size_t)255;
        return r;
    };
    __bf16* xb   = (__bf16*)take((size_t)T_TOK * HDIM * 2);
    __bf16* W1T  = (__bf16*)take((size_t)NEXP * DDIM * HDIM * 2);
    __bf16* W2T  = (__bf16*)take((size_t)NEXP * HDIM * DDIM * 2);
    __bf16* hmid = (__bf16*)take((size_t)CAPR * DDIM * 2);
    int*    row_token   = (int*)take((size_t)CAPR * 4);
    float*  row_gate    = (float*)take((size_t)CAPR * 4);
    int*    tile_expert = (int*)take((size_t)CAPT * 4);
    int*    top_e       = (int*)take((size_t)T_TOK * 2 * 4);
    float*  top_g       = (float*)take((size_t)T_TOK * 2 * 4);
    int*    counts      = (int*)take(NEXP * 4);
    int*    cursor      = (int*)take(NEXP * 4);
    __bf16* zrow        = (__bf16*)take((size_t)ZROWN * 2);

    const int TH = T_TOK * HDIM;

    // 1) init
    zero_out_kernel<<<(TH + 255) / 256, 256, 0, stream>>>(out, TH);
    init_meta_kernel<<<(CAPR + 255) / 256, 256, 0, stream>>>(row_token, tile_expert,
                                                             counts, cursor, zrow);

    // 2) bf16 staging (x direct; weights transposed so WMMA B-operand loads are contiguous)
    convert_x_kernel<<<(TH + 255) / 256, 256, 0, stream>>>(x, xb, TH);
    transpose_bf16_kernel<<<dim3(DDIM / 32, HDIM / 32, NEXP), dim3(32, 8), 0, stream>>>(W1, W1T, HDIM, DDIM);
    transpose_bf16_kernel<<<dim3(HDIM / 32, DDIM / 32, NEXP), dim3(32, 8), 0, stream>>>(W2, W2T, DDIM, HDIM);

    // 3) routing + top-2 dispatch lists
    routing_kernel<<<T_TOK / 8, 256, 0, stream>>>(x, rw, top_e, top_g, counts);
    scan_kernel<<<1, 32, 0, stream>>>(counts, cursor, tile_expert);
    scatter_kernel<<<T_TOK / 256, 256, 0, stream>>>(top_e, top_g, cursor, row_token, row_gate);

    // 4) grouped expert FFN on dispatched rows only (2*T rows vs reference's 8*T)
    gemm1_kernel<<<dim3(CAPT, DDIM / 512), 256, 0, stream>>>(xb, W1T, b1, hmid,
                                                             row_token, tile_expert, zrow);
    gemm2_kernel<<<dim3(CAPT, HDIM / 512), 256, 0, stream>>>(hmid, W2T, b2, out,
                                                             row_token, row_gate, tile_expert);
}
